// Attention_54984171323822
// MI455X (gfx1250) — matrix operand order repaired and checked by reading, not compiled, not run on hardware
//
#include <hip/hip_runtime.h>
#include <hip/hip_bf16.h>

// ---------- config ----------
#define BATCH   4
#define SEQ     2048
#define HIDDEN  1024
#define NHEADS  16
#define DH      64
#define ATTN    1024      // NHEADS * DH
#define QKVE    3072      // 3 * ATTN

typedef _Float16 h16;
typedef __attribute__((ext_vector_type(16))) _Float16 v16h;
typedef __attribute__((ext_vector_type(8)))  _Float16 v8h;
typedef __attribute__((ext_vector_type(8)))  float    v8f;
typedef __attribute__((ext_vector_type(4)))  int      v4i;

// ---------- async global->LDS support (gfx1250) ----------
#if defined(__has_builtin)
#  if __has_builtin(__builtin_amdgcn_global_load_async_to_lds_b128)
#    define HAVE_ASYNC_LDS 1
#  endif
#endif
#ifndef HAVE_ASYNC_LDS
#  define HAVE_ASYNC_LDS 0
#endif

#if HAVE_ASYNC_LDS && __has_builtin(__builtin_amdgcn_s_wait_asynccnt)
#  define ASYNC_WAIT() __builtin_amdgcn_s_wait_asynccnt(0)
#elif HAVE_ASYNC_LDS
#  define ASYNC_WAIT() asm volatile("s_wait_asynccnt 0" ::: "memory")
#else
#  define ASYNC_WAIT() ((void)0)
#endif

// builtin signature: (int4 AS1*, int4 AS3*, imm int offset, imm int cpol)
typedef __attribute__((address_space(1))) v4i as1_v4i;
typedef __attribute__((address_space(3))) v4i as3_v4i;

// Copy 4KB contiguous global -> LDS (one wave). 8 x b128 per lane.
__device__ inline void copy_k_chunk(const h16* gsrc, h16* ldst) {
    int lane = threadIdx.x & 31;
#if HAVE_ASYNC_LDS
    as1_v4i* g = (as1_v4i*)(gsrc + lane * 8);     // 16B per lane
    as3_v4i* l = (as3_v4i*)(ldst + lane * 8);
#pragma unroll
    for (int i = 0; i < 8; ++i)
        __builtin_amdgcn_global_load_async_to_lds_b128(g + i * 32, l + i * 32, 0, 0);
#else
#pragma unroll
    for (int i = 0; i < 8; ++i) {
        v8h t = *(const v8h*)(gsrc + lane * 8 + i * 256);
        *(v8h*)(ldst + lane * 8 + i * 256) = t;
    }
#endif
}

// Copy VT chunk: 64 rows (d), 32 halves each (64B), global row stride SEQ.
// LDS layout: [d][32] contiguous.  2 rows per lane, 4 x b128 per row.
__device__ inline void copy_vt_chunk(const h16* gsrc, h16* ldst) {
    int lane = threadIdx.x & 31;
#if HAVE_ASYNC_LDS
#pragma unroll
    for (int rr = 0; rr < 2; ++rr) {
        int r = lane * 2 + rr;
        as1_v4i* g = (as1_v4i*)(gsrc + (size_t)r * SEQ);
        as3_v4i* l = (as3_v4i*)(ldst + r * 32);
#pragma unroll
        for (int i = 0; i < 4; ++i)
            __builtin_amdgcn_global_load_async_to_lds_b128(g + i, l + i, 0, 0);
    }
#else
#pragma unroll
    for (int rr = 0; rr < 2; ++rr) {
        int r = lane * 2 + rr;
#pragma unroll
        for (int i = 0; i < 4; ++i) {
            v8h t = *(const v8h*)(gsrc + (size_t)r * SEQ + i * 8);
            *(v8h*)(ldst + r * 32 + i * 8) = t;
        }
    }
#endif
}

// ---------- WMMA fragment helpers (wave32, 16x16x32 f16) ----------
// A-matrix 16x32 (MxK), 16-bit layout (ISA 7.12.2):
//   lanes 0-15 : row M=lane, halves = K[0..7]  then K[16..23]
//   lanes 16-31: row M=lane-16, halves = K[8..15] then K[24..31]
__device__ inline v16h load_frag_a(const h16* p0, int ld) {
    int lane = threadIdx.x & 31;
    int row  = lane & 15;
    int kh   = (lane >> 4) << 3;        // 0 or 8
    const h16* p = p0 + (size_t)row * ld + kh;
    v8h lo = *(const v8h*)(p);
    v8h hi = *(const v8h*)(p + 16);
    return __builtin_shufflevector(lo, hi, 0,1,2,3,4,5,6,7,8,9,10,11,12,13,14,15);
}

// B-matrix 32x16 (KxN): column n of B contiguous over K at p0 + n*ld.
// lanes 0-15: col=lane, K=0..7 then 16..23; lanes 16-31: col=lane-16, K=8..15 then 24..31.
__device__ inline v16h load_frag_b(const h16* p0, int ld) {
    int lane = threadIdx.x & 31;
    int col  = lane & 15;
    int kr   = (lane >> 4) << 3;        // 0 or 8
    const h16* p = p0 + (size_t)col * ld + kr;
    v8h lo = *(const v8h*)(p);
    v8h hi = *(const v8h*)(p + 16);
    return __builtin_shufflevector(lo, hi, 0,1,2,3,4,5,6,7,8,9,10,11,12,13,14,15);
}

__device__ inline v8f wmma_f16(v16h a, v16h b, v8f c) {
    return __builtin_amdgcn_wmma_f32_16x16x32_f16(false, a, false, b, (short)0, c,
                                                  false, false);
}

// ---------- 1) f32 -> f16 convert ----------
__global__ void cvt_f32_f16(const float* __restrict__ in, h16* __restrict__ out, int n) {
    int i = blockIdx.x * blockDim.x + threadIdx.x;
    if (i < n) out[i] = (h16)in[i];
}

// ---------- 2) QKV projection: qkv[m,e] = x[m,:] . w_qkv[e,:] + b ----------
__global__ __launch_bounds__(32) void qkv_gemm(
    const h16* __restrict__ X, const h16* __restrict__ W,
    const float* __restrict__ bias,
    h16* __restrict__ Q, h16* __restrict__ K, h16* __restrict__ VT) {
    int tm = blockIdx.x;                 // 0..127  (64 rows of 8192)
    int tn = blockIdx.y;                 // 0..47   (64 cols of 3072)
    int lane = threadIdx.x;
    v8f acc[4][4] = {};
    const h16* xrow = X + (size_t)tm * 64 * HIDDEN;
    const h16* wrow = W + (size_t)tn * 64 * HIDDEN;
    for (int k0 = 0; k0 < HIDDEN; k0 += 32) {
        v16h a[4];
#pragma unroll
        for (int mt = 0; mt < 4; ++mt)
            a[mt] = load_frag_a(xrow + (size_t)mt * 16 * HIDDEN + k0, HIDDEN);
#pragma unroll
        for (int nt = 0; nt < 4; ++nt) {
            v16h b = load_frag_b(wrow + (size_t)nt * 16 * HIDDEN + k0, HIDDEN);
#pragma unroll
            for (int mt = 0; mt < 4; ++mt)
                acc[mt][nt] = wmma_f16(a[mt], b, acc[mt][nt]);
        }
    }
    int col = lane & 15;
    int rb  = (lane >> 4) << 3;          // 0 or 8
#pragma unroll
    for (int mt = 0; mt < 4; ++mt)
#pragma unroll
        for (int nt = 0; nt < 4; ++nt)
#pragma unroll
            for (int r = 0; r < 8; ++r) {
                int m = tm * 64 + mt * 16 + rb + r;      // global row (b*SEQ+n)
                int e = tn * 64 + nt * 16 + col;         // 0..3071
                h16 hv = (h16)(acc[mt][nt][r] + bias[e]);
                int b_  = m >> 11;                        // m / SEQ
                int n_  = m & (SEQ - 1);
                int wch = e >> 10;                        // 0=Q 1=K 2=V
                int eh  = e & (ATTN - 1);
                int h_  = eh >> 6;
                int d_  = eh & (DH - 1);
                size_t bh = (size_t)(b_ * NHEADS + h_);
                if (wch == 0)      Q[(bh * SEQ + n_) * DH + d_] = hv;
                else if (wch == 1) K[(bh * SEQ + n_) * DH + d_] = hv;
                else               VT[(bh * DH + d_) * SEQ + n_] = hv;
            }
}

// ---------- 3) flash attention with async double-buffered K/V in LDS ----------
__global__ __launch_bounds__(32) void attn_kernel(
    const h16* __restrict__ Q, const h16* __restrict__ K,
    const h16* __restrict__ VT, h16* __restrict__ AO) {
    __shared__ __align__(64) h16 s_k[2][32 * DH];     // [buf][key*64+d]   4KB each
    __shared__ __align__(64) h16 s_vt[2][DH * 32];    // [buf][d*32+key]   4KB each
    __shared__ __align__(32) float s_s[16][32];       // raw scores (f32)
    __shared__ __align__(32) h16   s_p[16][32];       // probabilities (f16)
    __shared__ float s_m[16], s_l[16], s_a[16];

    int bh   = blockIdx.x;               // 0..63
    int qt   = blockIdx.y;               // 0..127
    int lane = threadIdx.x;

    const h16* qb  = Q  + ((size_t)bh * SEQ + qt * 16) * DH;
    const h16* kb0 = K  + (size_t)bh * SEQ * DH;
    const h16* vtb = VT + (size_t)bh * DH * SEQ;

    v16h qa0 = load_frag_a(qb + 0,  DH);
    v16h qa1 = load_frag_a(qb + 32, DH);

    if (lane < 16) { s_m[lane] = -1e30f; s_l[lane] = 0.0f; }

    // prime buffer 0
    copy_k_chunk(kb0, s_k[0]);
    copy_vt_chunk(vtb, s_vt[0]);
    ASYNC_WAIT();
    __syncthreads();

    v8f o[4] = {};
    int col = lane & 15;
    int rb  = (lane >> 4) << 3;

    for (int kt = 0; kt < SEQ; kt += 32) {
        int cur = (kt >> 5) & 1;
        int nxt = cur ^ 1;
        // issue async fill of the next chunk while computing this one
        if (kt + 32 < SEQ) {
            copy_k_chunk(kb0 + (size_t)(kt + 32) * DH, s_k[nxt]);
            copy_vt_chunk(vtb + (kt + 32), s_vt[nxt]);
        }

        // ---- scores for 32 keys (two 16x16 tiles, K-dim = 64) from LDS ----
#pragma unroll
        for (int j = 0; j < 2; ++j) {
            v8f s = {};
            const h16* kb = &s_k[cur][(j * 16) * DH];
            s = wmma_f16(qa0, load_frag_b(kb + 0,  DH), s);
            s = wmma_f16(qa1, load_frag_b(kb + 32, DH), s);
#pragma unroll
            for (int r = 0; r < 8; ++r)
                s_s[rb + r][j * 16 + col] = s[r] * 0.125f;   // * Dh^-0.5
        }
        __syncthreads();

        // ---- online softmax: one query row per lane (lanes 0..15) ----
        if (lane < 16) {
            float mo = s_m[lane];
            float mx = mo;
#pragma unroll
            for (int i = 0; i < 32; ++i) mx = fmaxf(mx, s_s[lane][i]);
            float alpha = __expf(mo - mx);
            float sum = 0.0f;
#pragma unroll
            for (int i = 0; i < 32; ++i) {
                float p = __expf(s_s[lane][i] - mx);
                sum += p;
                s_p[lane][i] = (h16)p;
            }
            s_m[lane] = mx;
            s_l[lane] = s_l[lane] * alpha + sum;
            s_a[lane] = alpha;
        }
        __syncthreads();

        // ---- P as A-fragment from LDS, rescale O, accumulate P*V ----
        v16h pa;
        {
            int row = lane & 15;
            int kh  = (lane >> 4) << 3;          // 0 or 8
            v8h lo = *(const v8h*)&s_p[row][kh];
            v8h hi = *(const v8h*)&s_p[row][kh + 16];
            pa = __builtin_shufflevector(lo, hi, 0,1,2,3,4,5,6,7,8,9,10,11,12,13,14,15);
        }
#pragma unroll
        for (int t = 0; t < 4; ++t) {
#pragma unroll
            for (int r = 0; r < 8; ++r) o[t][r] *= s_a[rb + r];
            // B[k=key][n=dim]: column (dim) contiguous over keys in LDS VT tile
            const h16* vb = &s_vt[cur][(t * 16) * 32];
            o[t] = wmma_f16(pa, load_frag_b(vb, 32), o[t]);
        }

        // next chunk must be resident before we flip buffers
        ASYNC_WAIT();
        __syncthreads();
    }

    // ---- epilogue: normalize, store [B,N,ATTN] f16 for final GEMM ----
    int b_ = bh >> 4, h_ = bh & 15;
#pragma unroll
    for (int t = 0; t < 4; ++t)
#pragma unroll
        for (int r = 0; r < 8; ++r) {
            int qrow = qt * 16 + rb + r;
            float v = o[t][r] / s_l[rb + r];
            AO[((size_t)b_ * SEQ + qrow) * ATTN + h_ * DH + t * 16 + col] = (h16)v;
        }
}

// ---------- 4) output projection: out[m,e] = ao[m,:] . w_out[e,:] + b ----------
__global__ __launch_bounds__(32) void out_gemm(
    const h16* __restrict__ AO, const h16* __restrict__ W,
    const float* __restrict__ bias, float* __restrict__ OUT) {
    int tm = blockIdx.x;                 // 0..127
    int tn = blockIdx.y;                 // 0..15
    int lane = threadIdx.x;
    v8f acc[4][4] = {};
    const h16* arow = AO + (size_t)tm * 64 * ATTN;
    const h16* wrow = W  + (size_t)tn * 64 * ATTN;
    for (int k0 = 0; k0 < ATTN; k0 += 32) {
        v16h a[4];
#pragma unroll
        for (int mt = 0; mt < 4; ++mt)
            a[mt] = load_frag_a(arow + (size_t)mt * 16 * ATTN + k0, ATTN);
#pragma unroll
        for (int nt = 0; nt < 4; ++nt) {
            v16h b = load_frag_b(wrow + (size_t)nt * 16 * ATTN + k0, ATTN);
#pragma unroll
            for (int mt = 0; mt < 4; ++mt)
                acc[mt][nt] = wmma_f16(a[mt], b, acc[mt][nt]);
        }
    }
    int col = lane & 15;
    int rb  = (lane >> 4) << 3;
#pragma unroll
    for (int mt = 0; mt < 4; ++mt)
#pragma unroll
        for (int nt = 0; nt < 4; ++nt)
#pragma unroll
            for (int r = 0; r < 8; ++r) {
                int m = tm * 64 + mt * 16 + rb + r;
                int e = tn * 64 + nt * 16 + col;
                OUT[(size_t)m * HIDDEN + e] = acc[mt][nt][r] + bias[e];
            }
}

// ---------- host ----------
extern "C" void kernel_launch(void* const* d_in, const int* in_sizes, int n_in,
                              void* d_out, int out_size, void* d_ws, size_t ws_size,
                              hipStream_t stream) {
    const float* x      = (const float*)d_in[0];   // [4,2048,1024]
    const float* w_qkv  = (const float*)d_in[1];   // [3072,1024]
    const float* b_qkv  = (const float*)d_in[2];   // [3072]
    const float* w_out  = (const float*)d_in[3];   // [1024,1024]
    const float* b_out  = (const float*)d_in[4];   // [1024]
    float* out = (float*)d_out;

    const int NX  = BATCH * SEQ * HIDDEN;          // 8388608
    const int NWQ = QKVE * HIDDEN;                 // 3145728
    const int NWO = HIDDEN * ATTN;                 // 1048576
    const int NQK = BATCH * NHEADS * SEQ * DH;     // 8388608
    const int NAO = BATCH * SEQ * ATTN;            // 8388608

    char* ws = (char*)d_ws;
    size_t off = 0;
    auto carve = [&](size_t nhalf) -> h16* {
        h16* p = (h16*)(ws + off);
        off += ((nhalf * sizeof(h16) + 255) / 256) * 256;
        return p;
    };
    h16* x16   = carve(NX);
    h16* wq16  = carve(NWQ);
    h16* wo16  = carve(NWO);
    h16* q16   = carve(NQK);
    h16* k16   = carve(NQK);
    h16* vt16  = carve(NQK);
    h16* ao16  = carve(NAO);
    (void)ws_size; (void)in_sizes; (void)n_in; (void)out_size;

    cvt_f32_f16<<<(NX  + 255) / 256, 256, 0, stream>>>(x,     x16,  NX);
    cvt_f32_f16<<<(NWQ + 255) / 256, 256, 0, stream>>>(w_qkv, wq16, NWQ);
    cvt_f32_f16<<<(NWO + 255) / 256, 256, 0, stream>>>(w_out, wo16, NWO);

    qkv_gemm<<<dim3(128, 48), 32, 0, stream>>>(x16, wq16, b_qkv, q16, k16, vt16);
    attn_kernel<<<dim3(64, 128), 32, 0, stream>>>(q16, k16, vt16, ao16);
    out_gemm<<<dim3(128, 16), 32, 0, stream>>>(ao16, wo16, b_out, out);
}
